// MultiHeadLatentAttention_74517682585991
// MI455X (gfx1250) — matrix operand order repaired and checked by reading, not compiled, not run on hardware
//
#include <hip/hip_runtime.h>
#include <hip/hip_bf16.h>

// ---------------------------------------------------------------------------
// MLA forward for gfx1250 (MI455X).  Compute-bound (~230 GFLOP vs ~300MB HBM)
// => all GEMMs + attention through v_wmma_f32_16x16x32_f16 (f32 accumulate).
// All WMMA operands are pre-converted to f16 (weights transposed to K-major
// once) so GEMM staging is pure b128 global->LDS copies, no VALU in the loop.
// Workspace requirement: ~290 MB in d_ws.
// ---------------------------------------------------------------------------

typedef _Float16 f16;
typedef __attribute__((ext_vector_type(16))) _Float16 v16h;
typedef __attribute__((ext_vector_type(8)))  _Float16 v8h;
typedef __attribute__((ext_vector_type(4)))  _Float16 v4h;
typedef __attribute__((ext_vector_type(8)))  float    v8f;
typedef __attribute__((ext_vector_type(4)))  float    v4f;

union Frag { v16h v; v8h h[2]; };

#define WMMA(a, b, c) \
    __builtin_amdgcn_wmma_f32_16x16x32_f16(false, (a), false, (b), (short)0, (c), false, false)

#define MLA_B   2
#define MLA_T   2048
#define MLA_D   2048
#define MLA_H   16
#define MLA_NOPE 128
#define MLA_ROPE 64
#define MLA_QH  192          // NOPE + ROPE
#define MLA_QL  768          // H*QH/4
#define MLA_KVR 512
#define MLA_VD  128
#define ATT_SCALE 0.07216878364870323f   // 192^-0.5

// ---------------------------------------------------------------------------
// fp32 -> f16 elementwise convert (vectorized x4).  n4 = n/4.
// ---------------------------------------------------------------------------
__global__ __launch_bounds__(256) void cvt_f32_to_f16(
    const float* __restrict__ X, f16* __restrict__ Y, int n4) {
  int i = blockIdx.x * 256 + threadIdx.x;
  if (i < n4) {
    v4f x = *reinterpret_cast<const v4f*>(X + (size_t)i * 4);
    v4h y;
    y[0] = (f16)x[0]; y[1] = (f16)x[1]; y[2] = (f16)x[2]; y[3] = (f16)x[3];
    *reinterpret_cast<v4h*>(Y + (size_t)i * 4) = y;
  }
}

// ---------------------------------------------------------------------------
// Weight transpose + convert:  Y[n,k](f16) = X[k,n](f32).  LDS 32x32 tiles so
// both the global read and the global write are coalesced.  block (32,8).
// ---------------------------------------------------------------------------
__global__ __launch_bounds__(256) void transpose_cvt_f16(
    const float* __restrict__ X, f16* __restrict__ Y, int K, int N) {
  __shared__ f16 tile[32][33];
  const int k0 = blockIdx.y * 32, n0 = blockIdx.x * 32;
  const int tx = threadIdx.x, ty = threadIdx.y;
  for (int i = 0; i < 4; ++i) {
    int k = k0 + ty + i * 8, n = n0 + tx;
    f16 v = (f16)0.0f;
    if (k < K && n < N) v = (f16)X[(size_t)k * N + n];
    tile[ty + i * 8][tx] = v;
  }
  __syncthreads();
  for (int i = 0; i < 4; ++i) {
    int n = n0 + ty + i * 8, k = k0 + tx;
    if (n < N && k < K) Y[(size_t)n * K + k] = tile[tx][ty + i * 8];
  }
}

// ---------------------------------------------------------------------------
// GEMM:  C[M,N](f32) = A[M,K](f16) * Bt[N,K](f16, K-major).
// Block tile 128x256, 8 waves (2Mx4N), wave tile 64x64 = 4x4 16x16 frags.
// 16 WMMAs per wave per K-step vs 16 fragment b128 loads (1:1).
// M, K multiples of 128/32 (true for all calls); N guarded.
// ---------------------------------------------------------------------------
__global__ __launch_bounds__(256) void gemm_wmma_f16(
    const f16* __restrict__ A, const f16* __restrict__ Bt,
    float* __restrict__ C, int M, int N, int K) {
  __shared__ f16 smA[128][40];   // [m][k], stride 40 halves (16B-aligned rows)
  __shared__ f16 smB[256][40];   // [n][k]

  const int tid   = threadIdx.x;
  const int lane  = tid & 31;
  const int wave  = tid >> 5;
  const int bm    = blockIdx.y * 128;
  const int bn    = blockIdx.x * 256;
  const int wm    = (wave & 1) * 64;
  const int wn    = (wave >> 1) * 64;
  const int hlane = lane & 15;
  const int kbA   = (lane < 16) ? 0 : 8;    // ISA A-frag K split
  const int kbB   = (lane < 16) ? 0 : 16;   // unused: B frags load with kbA (same K split as A)
  const int rbase = (lane < 16) ? 0 : 8;    // ISA C-frag row split

  v8f acc[4][4];
  for (int i = 0; i < 4; ++i)
    for (int j = 0; j < 4; ++j)
      for (int r = 0; r < 8; ++r) acc[i][j][r] = 0.0f;

  for (int kt = 0; kt < K; kt += 32) {
    // Stage A tile 128x32: pure b128 copies.
    for (int i = 0; i < 2; ++i) {
      int idx8 = (tid + 256 * i) * 8;
      int r = idx8 >> 5, c = idx8 & 31;
      *reinterpret_cast<v8h*>(&smA[r][c]) =
          *reinterpret_cast<const v8h*>(A + (size_t)(bm + r) * K + kt + c);
    }
    // Stage B tile 256 rows (n) x 32 (k): pure b128 copies (K-major source).
    for (int i = 0; i < 4; ++i) {
      int idx8 = (tid + 256 * i) * 8;
      int n = idx8 >> 5, c = idx8 & 31;
      v8h v;
      for (int j = 0; j < 8; ++j) v[j] = (f16)0.0f;
      if (bn + n < N)
        v = *reinterpret_cast<const v8h*>(Bt + (size_t)(bn + n) * K + kt + c);
      *reinterpret_cast<v8h*>(&smB[n][c]) = v;
    }
    // Prefetch next K-slab into GL2 while we compute.
    if (kt + 32 < K) {
      __builtin_prefetch(A + (size_t)(bm + (tid >> 1)) * K + kt + 32 + (tid & 1) * 16, 0, 1);
      if (bn + tid < N)
        __builtin_prefetch(Bt + (size_t)(bn + tid) * K + kt + 32, 0, 1);
    }
    __syncthreads();

    Frag af[4], bf[4];
    for (int mf = 0; mf < 4; ++mf) {
      const f16* rp = &smA[wm + mf * 16 + hlane][0];
      af[mf].h[0] = *reinterpret_cast<const v8h*>(rp + kbA);
      af[mf].h[1] = *reinterpret_cast<const v8h*>(rp + 16 + kbA);
    }
    for (int nf = 0; nf < 4; ++nf) {
      const f16* rp = &smB[wn + nf * 16 + hlane][0];
      bf[nf].h[0] = *reinterpret_cast<const v8h*>(rp + kbA);
      bf[nf].h[1] = *reinterpret_cast<const v8h*>(rp + 16 + kbA);
    }
    for (int mf = 0; mf < 4; ++mf)
      for (int nf = 0; nf < 4; ++nf)
        acc[mf][nf] = WMMA(af[mf].v, bf[nf].v, acc[mf][nf]);
    __syncthreads();
  }

  for (int mf = 0; mf < 4; ++mf)
    for (int nf = 0; nf < 4; ++nf)
      for (int r = 0; r < 8; ++r) {
        int row = bm + wm + mf * 16 + rbase + r;
        int col = bn + wn + nf * 16 + hlane;
        if (col < N) C[(size_t)row * N + col] = acc[mf][nf][r];
      }
}

// ---------------------------------------------------------------------------
// Row-wise RMSNorm with dual outputs:
//   Yf (f32, may be null)  and  Yh (f16, may be null)
// ---------------------------------------------------------------------------
__global__ __launch_bounds__(256) void rmsnorm_rows(
    const float* __restrict__ X, int ldx, const float* __restrict__ g,
    float* __restrict__ Yf, f16* __restrict__ Yh, int ldy, int n) {
  __shared__ float red[8];
  const int row = blockIdx.x, tid = threadIdx.x;
  const float* x = X + (size_t)row * ldx;
  float ss = 0.0f;
  for (int c = tid; c < n; c += 256) { float v = x[c]; ss += v * v; }
  for (int off = 16; off > 0; off >>= 1) ss += __shfl_xor(ss, off, 32);
  if ((tid & 31) == 0) red[tid >> 5] = ss;
  __syncthreads();
  float tot = 0.0f;
  for (int i = 0; i < 8; ++i) tot += red[i];
  float sc = rsqrtf(tot / (float)n + 1e-6f);
  for (int c = tid; c < n; c += 256) {
    float v = x[c] * sc * g[c];
    if (Yf) Yf[(size_t)row * ldy + c] = v;
    if (Yh) Yh[(size_t)row * ldy + c] = (f16)v;
  }
}

// Copy k_rope slice (cols 512..575 of kvc) into its d_out slot.
__global__ void copy_krope(const float* __restrict__ kvc, float* __restrict__ out) {
  int i = blockIdx.x * 256 + threadIdx.x;     // B*T*64 elements
  int bt = i >> 6, d = i & 63;
  out[i] = kvc[(size_t)bt * 576 + 512 + d];
}

// ---------------------------------------------------------------------------
// Pack Q (B,T,H,192)f32 -> (B,H,T,192)f16 with RoPE on last 64 dims.
// One block (64 threads) per (b,t,h).
// ---------------------------------------------------------------------------
__global__ void pack_q_rope(const float* __restrict__ Qraw,
                            const int* __restrict__ pos, f16* __restrict__ Qf) {
  int idx = blockIdx.x;                 // b*T*H + t*H + h
  int h = idx & (MLA_H - 1);
  int t = (idx >> 4) & (MLA_T - 1);
  int b = idx >> 15;                    // T*H = 2^15
  const float* src = Qraw + (size_t)idx * MLA_QH;
  f16* dst = Qf + (((size_t)(b * MLA_H + h)) * MLA_T + t) * MLA_QH;
  int j = threadIdx.x;                  // 0..63
  dst[j]      = (f16)src[j];
  dst[j + 64] = (f16)src[j + 64];
  if (j < 32) {
    float p   = (float)pos[b * MLA_T + t];
    float inv = powf(10000.0f, -(float)(2 * j) / (float)MLA_ROPE);
    float ang = p * inv;
    float c = cosf(ang), s = sinf(ang);
    float x1 = src[128 + j], x2 = src[160 + j];
    dst[128 + j] = (f16)(x1 * c - x2 * s);
    dst[160 + j] = (f16)(x2 * c + x1 * s);
  }
}

// ---------------------------------------------------------------------------
// Pack K/V: kv (B,T,H,256)f32 + k_rope (B,T,64)f32 ->
//   Kf (B,H,T,192)f16 (nope | rope'd, rope broadcast over heads)
//   Vf (B,H,T,128)f16
// ---------------------------------------------------------------------------
__global__ void pack_kv_rope(const float* __restrict__ KVraw,
                             const float* __restrict__ krope,
                             const int* __restrict__ pos,
                             f16* __restrict__ Kf, f16* __restrict__ Vf) {
  int idx = blockIdx.x;
  int h = idx & (MLA_H - 1);
  int t = (idx >> 4) & (MLA_T - 1);
  int b = idx >> 15;
  const float* src = KVraw + (size_t)idx * 256;
  size_t obase = ((size_t)(b * MLA_H + h)) * MLA_T + t;
  f16* kd = Kf + obase * MLA_QH;
  f16* vd = Vf + obase * MLA_VD;
  int j = threadIdx.x;                  // 0..63
  kd[j]      = (f16)src[j];
  kd[j + 64] = (f16)src[j + 64];
  vd[j]      = (f16)src[128 + j];
  vd[j + 64] = (f16)src[192 + j];
  if (j < 32) {
    const float* kr = krope + ((size_t)(b * MLA_T + t)) * MLA_ROPE;
    float p   = (float)pos[b * MLA_T + t];
    float inv = powf(10000.0f, -(float)(2 * j) / (float)MLA_ROPE);
    float ang = p * inv;
    float c = cosf(ang), s = sinf(ang);
    float x1 = kr[j], x2 = kr[32 + j];
    kd[128 + j] = (f16)(x1 * c - x2 * s);
    kd[160 + j] = (f16)(x2 * c + x1 * s);
  }
}

// ---------------------------------------------------------------------------
// Causal flash attention.  Grid: (T/64, B*H).  Block: 128 threads = 4 waves.
// Wave w owns query rows [qt+16w, qt+16w+16).  Online softmax, WMMA for
// S = Q K^T (24 wmma / key tile) and O += P V (16 wmma / key tile).
// Output written directly in (B, T, H*V_DIM) f16 layout for the final GEMM.
// ---------------------------------------------------------------------------
__global__ __launch_bounds__(128) void mla_flash_attn(
    const f16* __restrict__ Q, const f16* __restrict__ Kt,
    const f16* __restrict__ V, f16* __restrict__ O, int T) {
  __shared__ f16 smK[64][200];      // 64 keys x 192 dims
  __shared__ f16 smV[128][72];      // V transposed: [d][key]
  __shared__ f16 smP[4][16][72];    // per-wave P scratch (A-frag relayout)

  const int tid   = threadIdx.x;
  const int lane  = tid & 31;
  const int wave  = tid >> 5;
  const int hlane = lane & 15;
  const int kbA   = (lane < 16) ? 0 : 8;
  const int kbB   = (lane < 16) ? 0 : 16;
  const int rbase = (lane < 16) ? 0 : 8;
  const int bh    = blockIdx.y;
  const int b     = bh >> 4;
  const int h     = bh & 15;
  const int qt    = blockIdx.x * 64;
  const int qrow  = qt + wave * 16 + hlane;

  // Preload Q fragments (16 rows x 192 dims = 6 A-frags), straight from global.
  Frag qf[6];
  {
    const f16* qp = Q + ((size_t)bh * T + qrow) * MLA_QH;
    for (int kc = 0; kc < 6; ++kc) {
      qf[kc].h[0] = *reinterpret_cast<const v8h*>(qp + kc * 32 + kbA);
      qf[kc].h[1] = *reinterpret_cast<const v8h*>(qp + kc * 32 + 16 + kbA);
    }
  }

  float m[8], l[8];
  v8f o[8];
  for (int r = 0; r < 8; ++r) { m[r] = -1e30f; l[r] = 0.0f; }
  for (int dc = 0; dc < 8; ++dc)
    for (int r = 0; r < 8; ++r) o[dc][r] = 0.0f;

  for (int kt = 0; kt <= qt; kt += 64) {
    // ---- cooperative stage: K tile (64x192) and V tile transposed (128x64)
    for (int i = 0; i < 12; ++i) {
      int idx8 = (tid + 128 * i) * 8;            // 12288 halves
      int key = idx8 / 192, d = idx8 % 192;
      v8h kv8 = *reinterpret_cast<const v8h*>(
          Kt + ((size_t)bh * T + kt + key) * MLA_QH + d);
      *reinterpret_cast<v8h*>(&smK[key][d]) = kv8;
    }
    for (int i = 0; i < 8; ++i) {
      int idx8 = (tid + 128 * i) * 8;            // 8192 halves
      int key = idx8 >> 7, d0 = idx8 & 127;
      v8h vv = *reinterpret_cast<const v8h*>(
          V + ((size_t)bh * T + kt + key) * MLA_VD + d0);
      for (int j = 0; j < 8; ++j) smV[d0 + j][key] = vv[j];
    }
    __syncthreads();

    // ---- S = Q * K^T  (16 q-rows x 64 keys per wave)
    v8f s[4];
    for (int kn = 0; kn < 4; ++kn)
      for (int r = 0; r < 8; ++r) s[kn][r] = 0.0f;
    for (int kn = 0; kn < 4; ++kn) {
      const f16* kp = &smK[kn * 16 + hlane][0];
      for (int kc = 0; kc < 6; ++kc) {
        Frag bf;
        bf.h[0] = *reinterpret_cast<const v8h*>(kp + kc * 32 + kbA);
        bf.h[1] = *reinterpret_cast<const v8h*>(kp + kc * 32 + 16 + kbA);
        s[kn] = WMMA(qf[kc].v, bf.v, s[kn]);
      }
    }

    // ---- scale + causal mask (only the diagonal tile needs masking)
    const bool diag = (kt == qt);
    for (int kn = 0; kn < 4; ++kn)
      for (int r = 0; r < 8; ++r) {
        float v = s[kn][r] * ATT_SCALE;
        if (diag) {
          int col = kt + kn * 16 + hlane;
          int row = qt + wave * 16 + rbase + r;
          if (col > row) v = -1e30f;
        }
        s[kn][r] = v;
      }

    // ---- online softmax (row reductions across the 16-lane half-groups)
    for (int r = 0; r < 8; ++r) {
      float mx = fmaxf(fmaxf(s[0][r], s[1][r]), fmaxf(s[2][r], s[3][r]));
      for (int off = 1; off < 16; off <<= 1) mx = fmaxf(mx, __shfl_xor(mx, off, 32));
      float mn   = fmaxf(m[r], mx);
      float corr = __expf(m[r] - mn);
      float ps = 0.0f;
      for (int kn = 0; kn < 4; ++kn) {
        float p = __expf(s[kn][r] - mn);
        smP[wave][rbase + r][kn * 16 + hlane] = (f16)p;
        ps += p;
      }
      for (int off = 1; off < 16; off <<= 1) ps += __shfl_xor(ps, off, 32);
      l[r] = l[r] * corr + ps;
      m[r] = mn;
      for (int dc = 0; dc < 8; ++dc) o[dc][r] *= corr;
    }
    __syncthreads();   // make P visible across the wave's lanes

    // ---- O += P * V
    Frag pf[2];
    {
      const f16* pp = &smP[wave][hlane][0];
      for (int kc = 0; kc < 2; ++kc) {
        pf[kc].h[0] = *reinterpret_cast<const v8h*>(pp + kc * 32 + kbA);
        pf[kc].h[1] = *reinterpret_cast<const v8h*>(pp + kc * 32 + 16 + kbA);
      }
    }
    for (int dc = 0; dc < 8; ++dc) {
      const f16* vp = &smV[dc * 16 + hlane][0];
      for (int kc = 0; kc < 2; ++kc) {
        Frag bf;
        bf.h[0] = *reinterpret_cast<const v8h*>(vp + kc * 32 + kbA);
        bf.h[1] = *reinterpret_cast<const v8h*>(vp + kc * 32 + 16 + kbA);
        o[dc] = WMMA(pf[kc].v, bf.v, o[dc]);
      }
    }
    __syncthreads();   // protect smK/smV before next stage
  }

  // ---- normalize + write (B, T, H*V_DIM) f16 for the output projection GEMM
  for (int r = 0; r < 8; ++r) {
    float inv = 1.0f / l[r];
    int trow = qt + wave * 16 + rbase + r;
    for (int dc = 0; dc < 8; ++dc) {
      int col = dc * 16 + hlane;
      O[((size_t)(b * T + trow)) * (MLA_H * MLA_VD) + h * MLA_VD + col] =
          (f16)(o[dc][r] * inv);
    }
  }
}

// ---------------------------------------------------------------------------
// Host-side orchestration
// ---------------------------------------------------------------------------
extern "C" void kernel_launch(void* const* d_in, const int* in_sizes, int n_in,
                              void* d_out, int out_size, void* d_ws, size_t ws_size,
                              hipStream_t stream) {
  (void)in_sizes; (void)n_in; (void)out_size; (void)ws_size;
  const float* hidden    = (const float*)d_in[0];
  const int*   positions = (const int*)d_in[1];
  const float* w_qa      = (const float*)d_in[2];
  const float* g_qa      = (const float*)d_in[3];
  const float* w_qb      = (const float*)d_in[4];
  const float* w_kva     = (const float*)d_in[5];
  const float* g_kva     = (const float*)d_in[6];
  const float* w_kvb     = (const float*)d_in[7];
  const float* w_o       = (const float*)d_in[8];

  const size_t BT = (size_t)MLA_B * MLA_T;         // 4096
  float* out     = (float*)d_out;                  // (B,T,D)
  float* out_ckv = out + BT * MLA_D;               // (B,T,512)
  float* out_kr  = out_ckv + BT * MLA_KVR;         // (B,T,64)

  char* ws = (char*)d_ws;
  size_t off = 0;
  auto alloc = [&](size_t bytes) -> void* {
    void* p = ws + off;
    off += (bytes + 255) & ~(size_t)255;
    return p;
  };
  f16*   hidden_h = (f16*)alloc(BT * MLA_D * 2);                   // 16.8 MB
  f16*   wqa_t    = (f16*)alloc((size_t)MLA_QL * MLA_D * 2);       //  3.1 MB
  f16*   wkva_t   = (f16*)alloc((size_t)576 * MLA_D * 2);          //  2.4 MB
  f16*   wqb_t    = (f16*)alloc((size_t)MLA_H * MLA_QH * MLA_QL * 2); // 4.7 MB
  f16*   wkvb_t   = (f16*)alloc((size_t)MLA_H * 256 * MLA_KVR * 2);   // 4.2 MB
  f16*   wo_t     = (f16*)alloc((size_t)MLA_D * MLA_H * MLA_VD * 2);  // 8.4 MB
  float* qlora    = (float*)alloc(BT * MLA_QL * 4);                // 12.6 MB
  f16*   qlora_h  = (f16*)alloc(BT * MLA_QL * 2);                  //  6.3 MB
  float* kvc      = (float*)alloc(BT * (MLA_KVR + MLA_ROPE) * 4);  //  9.4 MB
  f16*   ckv_h    = (f16*)alloc(BT * MLA_KVR * 2);                 //  4.2 MB
  float* qraw     = (float*)alloc(BT * MLA_H * MLA_QH * 4);        // 50.3 MB
  float* kvraw    = (float*)alloc(BT * MLA_H * 256 * 4);           // 67.1 MB
  f16*   qf       = (f16*)alloc(BT * MLA_H * MLA_QH * 2);          // 25.2 MB
  f16*   kf       = (f16*)alloc(BT * MLA_H * MLA_QH * 2);          // 25.2 MB
  f16*   vf       = (f16*)alloc(BT * MLA_H * MLA_VD * 2);          // 16.8 MB
  f16*   attn_h   = (f16*)alloc(BT * MLA_H * MLA_VD * 2);          // 16.8 MB

  dim3 blk256(256);
  dim3 tblk(32, 8);

  // One-time operand preparation (f16 activations, K-major f16 weights).
  cvt_f32_to_f16<<<(int)(BT * MLA_D / 4 / 256), blk256, 0, stream>>>(
      hidden, hidden_h, (int)(BT * MLA_D / 4));
  transpose_cvt_f16<<<dim3(MLA_QL / 32, MLA_D / 32), tblk, 0, stream>>>(
      w_qa, wqa_t, MLA_D, MLA_QL);
  transpose_cvt_f16<<<dim3(576 / 32, MLA_D / 32), tblk, 0, stream>>>(
      w_kva, wkva_t, MLA_D, 576);
  transpose_cvt_f16<<<dim3(MLA_H * MLA_QH / 32, MLA_QL / 32), tblk, 0, stream>>>(
      w_qb, wqb_t, MLA_QL, MLA_H * MLA_QH);
  transpose_cvt_f16<<<dim3(MLA_H * 256 / 32, MLA_KVR / 32), tblk, 0, stream>>>(
      w_kvb, wkvb_t, MLA_KVR, MLA_H * 256);
  transpose_cvt_f16<<<dim3(MLA_D / 32, MLA_H * MLA_VD / 32), tblk, 0, stream>>>(
      w_o, wo_t, MLA_H * MLA_VD, MLA_D);

  // q_lora = hidden @ w_qa ; rms -> f16
  gemm_wmma_f16<<<dim3((MLA_QL + 255) / 256, BT / 128), blk256, 0, stream>>>(
      hidden_h, wqa_t, qlora, (int)BT, MLA_QL, MLA_D);
  rmsnorm_rows<<<(int)BT, 256, 0, stream>>>(
      qlora, MLA_QL, g_qa, (float*)nullptr, qlora_h, MLA_QL, MLA_QL);
  // kvc = hidden @ w_kva ; rms(:512) -> out_ckv(f32) + ckv_h(f16) ; rope slice
  gemm_wmma_f16<<<dim3((576 + 255) / 256, BT / 128), blk256, 0, stream>>>(
      hidden_h, wkva_t, kvc, (int)BT, 576, MLA_D);
  rmsnorm_rows<<<(int)BT, 256, 0, stream>>>(
      kvc, 576, g_kva, out_ckv, ckv_h, MLA_KVR, MLA_KVR);
  copy_krope<<<(int)(BT * MLA_ROPE / 256), 256, 0, stream>>>(kvc, out_kr);
  // q = rms(q_lora) @ w_qb
  gemm_wmma_f16<<<dim3(MLA_H * MLA_QH / 256, BT / 128), blk256, 0, stream>>>(
      qlora_h, wqb_t, qraw, (int)BT, MLA_H * MLA_QH, MLA_QL);
  // kv = new_c_kv @ w_kvb
  gemm_wmma_f16<<<dim3(MLA_H * 256 / 256, BT / 128), blk256, 0, stream>>>(
      ckv_h, wkvb_t, kvraw, (int)BT, MLA_H * 256, MLA_KVR);
  // pack + RoPE -> f16 (B,H,T,dh)
  pack_q_rope<<<(int)(BT * MLA_H), 64, 0, stream>>>(qraw, positions, qf);
  pack_kv_rope<<<(int)(BT * MLA_H), 64, 0, stream>>>(kvraw, out_kr, positions, kf, vf);
  // flash attention -> f16 (B,T,H*V_DIM)
  mla_flash_attn<<<dim3(MLA_T / 64, MLA_B * MLA_H), 128, 0, stream>>>(
      qf, kf, vf, attn_h, MLA_T);
  // output projection
  gemm_wmma_f16<<<dim3(MLA_D / 256, BT / 128), blk256, 0, stream>>>(
      attn_h, wo_t, out, (int)BT, MLA_D, MLA_H * MLA_VD);
}
